// Encoder_16595753632210
// MI455X (gfx1250) — matrix operand order repaired and checked by reading, not compiled, not run on hardware
//
#include <hip/hip_runtime.h>

#define PAD      0
#define N_LAYERS 6
#define NHEADS   8
#define DMODEL   512
#define DFF      2048
#define BATCH    8
#define SEQ      512
#define DH       64            // DMODEL / NHEADS
#define MROWS    (BATCH*SEQ)   // 4096
#define QT       32            // q rows per attention block

typedef __attribute__((ext_vector_type(16))) _Float16 v16h;
typedef __attribute__((ext_vector_type(8)))  float    v8f;

// ---------------------------------------------------------------------------
// Embedding + positional encoding (faithful to the reference's odd-index form)
// ---------------------------------------------------------------------------
__global__ void embed_kernel(const int* __restrict__ input,
                             const float* __restrict__ table,
                             float* __restrict__ x) {
  const size_t idx = (size_t)blockIdx.x * blockDim.x + threadIdx.x;
  if (idx >= (size_t)MROWS * DMODEL) return;
  const int d   = (int)(idx % DMODEL);
  const int row = (int)(idx / DMODEL);
  const int s   = row % SEQ;
  const int tok = input[row];
  const float emb = (tok == PAD) ? 0.f : table[(size_t)tok * DMODEL + d];
  const float pos = (float)s;
  float pe;
  if ((d & 1) == 0) {
    pe = __sinf(pos * __powf(10000.0f, -2.0f * (float)d / (float)DMODEL));
  } else {
    pe = __cosf(pos / (10000.0f * (2.0f * (float)d / (float)DMODEL)));
  }
  x[idx] = emb + pe;
}

// ---------------------------------------------------------------------------
// WMMA GEMM: C[M,N] = A[M,K] * B[K,N] (+bias) (+relu).  fp32 in/out, f16
// operands, f32 accumulate.  Block tile 128x128, K-step 32, 256 threads =
// 8 waves as 4(M) x 2(N); wave tile 32x64 (2x4 fragments).  Register
// double-buffering: tile k+1 is loaded from global into VGPRs while the
// WMMAs for tile k execute out of LDS.  __launch_bounds__(256,2) = one
// 8-wave block per WGP (2 waves/SIMD) so the ~150 live VGPRs fit with no
// scratch spills in the inner loop.
// ---------------------------------------------------------------------------
#define LDSP 40   // padded half-stride (32 data + 8 pad) to dodge bank conflicts

__device__ __forceinline__ void stage_regs(const float* __restrict__ A,
                                           const float* __restrict__ B,
                                           int K, int N, int m0, int n0, int k0,
                                           int tid, float4* ra, float4* rb) {
#pragma unroll
  for (int p = 0; p < 4; ++p) {
    const int idx = p * 256 + tid;
    ra[p] = *(const float4*)&A[(size_t)(m0 + (idx >> 3)) * K + k0 + (idx & 7) * 4];
    rb[p] = *(const float4*)&B[(size_t)(k0 + (idx >> 5)) * N + n0 + (idx & 31) * 4];
  }
}

__device__ __forceinline__ void regs_to_lds(_Float16* As, _Float16* Bs, int tid,
                                            const float4* ra, const float4* rb) {
#pragma unroll
  for (int p = 0; p < 4; ++p) {
    const int idx = p * 256 + tid;
    {
      const int row = idx >> 3;
      const int c4  = (idx & 7) * 4;
      _Float16* d = &As[row * LDSP + c4];
      d[0] = (_Float16)ra[p].x; d[1] = (_Float16)ra[p].y;
      d[2] = (_Float16)ra[p].z; d[3] = (_Float16)ra[p].w;
    }
    {
      const int kk = idx >> 5;
      const int n4 = (idx & 31) * 4;
      Bs[(n4 + 0) * LDSP + kk] = (_Float16)rb[p].x;
      Bs[(n4 + 1) * LDSP + kk] = (_Float16)rb[p].y;
      Bs[(n4 + 2) * LDSP + kk] = (_Float16)rb[p].z;
      Bs[(n4 + 3) * LDSP + kk] = (_Float16)rb[p].w;
    }
  }
}

__global__ void __launch_bounds__(256, 2)
gemm_kernel(const float* __restrict__ A,
            const float* __restrict__ B,
            float* __restrict__ C,
            const float* __restrict__ bias,
            int M, int N, int K, int relu) {
  __shared__ _Float16 As[128 * LDSP];   // [row][k]   row-major
  __shared__ _Float16 Bs[128 * LDSP];   // [n][k]     transposed

  const int tid  = threadIdx.x;
  const int lane = tid & 31;
  const int wave = tid >> 5;
  const int l15  = lane & 15;
  const int g    = lane >> 4;          // lane half-group
  const int mw   = wave & 3;           // 0..3 -> 32-row band
  const int nw   = wave >> 2;          // 0..1 -> 64-col band
  const int m0   = blockIdx.y * 128;
  const int n0   = blockIdx.x * 128;

  v8f acc[2][4];
#pragma unroll
  for (int i = 0; i < 2; ++i)
#pragma unroll
    for (int j = 0; j < 4; ++j) acc[i][j] = (v8f){};

  float4 ra[4], rb[4];
  stage_regs(A, B, K, N, m0, n0, 0, tid, ra, rb);

  for (int k0 = 0; k0 < K; k0 += 32) {
    regs_to_lds(As, Bs, tid, ra, rb);
    __syncthreads();

    if (k0 + 32 < K) {   // overlap next tile's global loads with the WMMAs
      __builtin_prefetch(&A[(size_t)(m0 + (tid >> 3)) * K + k0 + 64], 0, 1);
      stage_regs(A, B, K, N, m0, n0, k0 + 32, tid, ra, rb);
    }

    v16h afr[2];
#pragma unroll
    for (int mi = 0; mi < 2; ++mi) {
      const _Float16* ap = &As[(mw * 32 + mi * 16 + l15) * LDSP];
#pragma unroll
      for (int j = 0; j < 8; ++j) {
        afr[mi][j]     = ap[g * 16 + j];
        afr[mi][8 + j] = ap[g * 16 + 8 + j];
      }
    }
#pragma unroll
    for (int ni = 0; ni < 4; ++ni) {
      const _Float16* bp = &Bs[(nw * 64 + ni * 16 + l15) * LDSP + g * 16];
      v16h bfr;
#pragma unroll
      for (int j = 0; j < 16; ++j) bfr[j] = bp[j];
#pragma unroll
      for (int mi = 0; mi < 2; ++mi)
        acc[mi][ni] = __builtin_amdgcn_wmma_f32_16x16x32_f16(
            false, afr[mi], false, bfr, (short)0, acc[mi][ni], false, false);
    }
    __syncthreads();
  }

  // --- epilogue: bias / relu / store ---
#pragma unroll
  for (int ni = 0; ni < 4; ++ni) {
    const int col = n0 + nw * 64 + ni * 16 + l15;
    const float bv = bias ? bias[col] : 0.f;
#pragma unroll
    for (int mi = 0; mi < 2; ++mi) {
#pragma unroll
      for (int r = 0; r < 8; ++r) {
        const int rowg = m0 + mw * 32 + mi * 16 + r + 8 * g;
        float v = acc[mi][ni][r] + bv;
        if (relu) v = fmaxf(v, 0.f);
        C[(size_t)rowg * N + col] = v;
      }
    }
  }
}

// ---------------------------------------------------------------------------
// Fused attention: per (b,h, 32 q-rows).  Scores tile 32x512 in LDS, WMMA for
// Q*K^T and P*V, key-pad mask + softmax in LDS.  Softmax writes a normalized
// f16 probability tile (sP); V chunks are cooperatively staged transposed in
// LDS so P*V operands are contiguous ds_load_b128 fragments.
// Dynamic LDS: sS f32[32][520] | sP f16[32][528] | sQ f16[32][72]
//              | sVt f16[64][40] | sTok i32[512] | sRed f32[32][8]  (~110 KB)
// ---------------------------------------------------------------------------
#define SSTR (SEQ + 8)    // f32 score row stride
#define PSTR (SEQ + 16)   // f16 prob row stride (keeps 16B alignment)

__global__ void __launch_bounds__(256, 2)
attn_kernel(const float* __restrict__ qbuf,
            const float* __restrict__ kbuf,
            const float* __restrict__ vbuf,
            const int* __restrict__ tokens,
            float* __restrict__ outbuf) {
  extern __shared__ char smem_raw[];
  float*    sS   = (float*)smem_raw;                               // [QT][SSTR]
  _Float16* sP   = (_Float16*)(smem_raw + QT * SSTR * 4);          // [QT][PSTR]
  _Float16* sQ   = sP + QT * PSTR;                                 // [QT][DH+8]
  _Float16* sVt  = sQ + QT * (DH + 8);                             // [DH][40]
  int*      sTok = (int*)(sVt + DH * 40);                          // [SEQ]
  float*    sRed = (float*)(sTok + SEQ);                           // [QT][8]

  const int bh  = blockIdx.x;
  const int b   = bh / NHEADS;
  const int h   = bh % NHEADS;
  const int q0  = blockIdx.y * QT;

  const int tid  = threadIdx.x;
  const int lane = tid & 31;
  const int wave = tid >> 5;
  const int l15  = lane & 15;
  const int g    = lane >> 4;

  for (int i = tid; i < SEQ; i += 256) sTok[i] = tokens[b * SEQ + i];
  for (int i = tid; i < QT * DH; i += 256) {
    const int r = i >> 6;            // / DH
    const int c = i & (DH - 1);
    sQ[r * (DH + 8) + c] =
        (_Float16)qbuf[((size_t)(b * SEQ + q0 + r)) * DMODEL + h * DH + c];
  }
  __syncthreads();

  // ---- scores: S = (Q K^T) * 1/sqrt(dh), key-pad mask ----
  {
    const int mw = wave & 1;         // 16 q-rows
    const int nw = wave >> 1;        // 128-key strip
    for (int kt = 0; kt < 8; ++kt) {
      const int n0 = nw * 128 + kt * 16;
      v8f acc = (v8f){};
#pragma unroll
      for (int ks = 0; ks < 2; ++ks) {
        const _Float16* ap = &sQ[(mw * 16 + l15) * (DH + 8) + ks * 32];
        v16h a;
#pragma unroll
        for (int j = 0; j < 8; ++j) {
          a[j]     = ap[g * 16 + j];
          a[8 + j] = ap[g * 16 + 8 + j];
        }
        // K's [S,D] layout is exactly the B^T fragment layout: contiguous
        const float* kp =
            &kbuf[((size_t)(b * SEQ + n0 + l15)) * DMODEL + h * DH + ks * 32 + g * 16];
        const float4 k0v = *(const float4*)(kp + 0);
        const float4 k1v = *(const float4*)(kp + 4);
        const float4 k2v = *(const float4*)(kp + 8);
        const float4 k3v = *(const float4*)(kp + 12);
        v16h bfr;
        bfr[0]  = (_Float16)k0v.x; bfr[1]  = (_Float16)k0v.y;
        bfr[2]  = (_Float16)k0v.z; bfr[3]  = (_Float16)k0v.w;
        bfr[4]  = (_Float16)k1v.x; bfr[5]  = (_Float16)k1v.y;
        bfr[6]  = (_Float16)k1v.z; bfr[7]  = (_Float16)k1v.w;
        bfr[8]  = (_Float16)k2v.x; bfr[9]  = (_Float16)k2v.y;
        bfr[10] = (_Float16)k2v.z; bfr[11] = (_Float16)k2v.w;
        bfr[12] = (_Float16)k3v.x; bfr[13] = (_Float16)k3v.y;
        bfr[14] = (_Float16)k3v.z; bfr[15] = (_Float16)k3v.w;
        acc = __builtin_amdgcn_wmma_f32_16x16x32_f16(
            false, a, false, bfr, (short)0, acc, false, false);
      }
      const int col = n0 + l15;
      const bool masked = (sTok[col] == PAD);
#pragma unroll
      for (int r = 0; r < 8; ++r) {
        const int row = mw * 16 + r + 8 * g;
        sS[row * SSTR + col] = masked ? -1e9f : acc[r] * 0.125f;
      }
    }
  }
  __syncthreads();

  // ---- softmax over keys (8 threads per row, 64 cols each); emit f16 P ----
  {
    const int row = tid >> 3;
    const int seg = tid & 7;
    float* srow = &sS[row * SSTR];
    float lmax = -3.4e38f;
    for (int c = seg * 64; c < seg * 64 + 64; ++c) lmax = fmaxf(lmax, srow[c]);
    sRed[row * 8 + seg] = lmax;
    __syncthreads();
    float rmax = -3.4e38f;
#pragma unroll
    for (int j = 0; j < 8; ++j) rmax = fmaxf(rmax, sRed[row * 8 + j]);
    __syncthreads();
    float lsum = 0.f;
    for (int c = seg * 64; c < seg * 64 + 64; ++c) {
      const float e = __expf(srow[c] - rmax);
      srow[c] = e;
      lsum += e;
    }
    sRed[row * 8 + seg] = lsum;
    __syncthreads();
    float rsum = 0.f;
#pragma unroll
    for (int j = 0; j < 8; ++j) rsum += sRed[row * 8 + j];
    const float rinv = 1.f / rsum;
    _Float16* prow = &sP[row * PSTR];
    for (int c = seg * 64; c < seg * 64 + 64; ++c)
      prow[c] = (_Float16)(srow[c] * rinv);
  }

  // ---- O = P V  (wave: 16 q-rows x 16 dh-cols); V staged via LDS ----
  {
    const int mw = wave & 1;         // 16 q-rows
    const int nw = wave >> 1;        // 16-wide dh strip
    v8f oacc = (v8f){};
    for (int kt = 0; kt < 16; ++kt) {
      const int k0 = kt * 32;
      __syncthreads();   // previous chunk's reads done (and P visible at kt=0)
      {                  // stage V[k0..k0+31][dh] -> sVt[dh][key] as f16
        const int key = tid >> 3;
        const int c8  = (tid & 7) * 8;
        const float* vp =
            &vbuf[((size_t)(b * SEQ + k0 + key)) * DMODEL + h * DH + c8];
        const float4 v0 = *(const float4*)vp;
        const float4 v1 = *(const float4*)(vp + 4);
        sVt[(c8 + 0) * 40 + key] = (_Float16)v0.x;
        sVt[(c8 + 1) * 40 + key] = (_Float16)v0.y;
        sVt[(c8 + 2) * 40 + key] = (_Float16)v0.z;
        sVt[(c8 + 3) * 40 + key] = (_Float16)v0.w;
        sVt[(c8 + 4) * 40 + key] = (_Float16)v1.x;
        sVt[(c8 + 5) * 40 + key] = (_Float16)v1.y;
        sVt[(c8 + 6) * 40 + key] = (_Float16)v1.z;
        sVt[(c8 + 7) * 40 + key] = (_Float16)v1.w;
      }
      __syncthreads();

      const _Float16* pp = &sP[(mw * 16 + l15) * PSTR + k0];
      v16h a;
#pragma unroll
      for (int j = 0; j < 8; ++j) {
        a[j]     = pp[g * 16 + j];
        a[8 + j] = pp[g * 16 + 8 + j];
      }
      const _Float16* vp2 = &sVt[(nw * 16 + l15) * 40 + g * 16];
      v16h bfr;
#pragma unroll
      for (int j = 0; j < 16; ++j) bfr[j] = vp2[j];
      oacc = __builtin_amdgcn_wmma_f32_16x16x32_f16(
          false, a, false, bfr, (short)0, oacc, false, false);
    }
#pragma unroll
    for (int r = 0; r < 8; ++r) {
      const int rr = q0 + mw * 16 + r + 8 * g;
      outbuf[((size_t)(b * SEQ + rr)) * DMODEL + h * DH + nw * 16 + l15] = oacc[r];
    }
  }
}

// ---------------------------------------------------------------------------
// out = LayerNorm(f + res) * gamma + beta, times non-pad mask.  One block/row.
// ---------------------------------------------------------------------------
__global__ void add_ln_kernel(const float* __restrict__ f,
                              const float* __restrict__ res,
                              const float* __restrict__ gamma,
                              const float* __restrict__ beta,
                              const int* __restrict__ tokens,
                              float* __restrict__ out) {
  __shared__ float red[256];
  const int row = blockIdx.x;
  const int tid = threadIdx.x;
  const float mask = (tokens[row] == PAD) ? 0.f : 1.f;
  const size_t base = (size_t)row * DMODEL;
  const float x0 = f[base + tid]       + res[base + tid];
  const float x1 = f[base + tid + 256] + res[base + tid + 256];
  red[tid] = x0 + x1;
  __syncthreads();
  for (int s = 128; s > 0; s >>= 1) {
    if (tid < s) red[tid] += red[tid + s];
    __syncthreads();
  }
  const float mu = red[0] * (1.0f / DMODEL);
  __syncthreads();
  const float d0 = x0 - mu, d1 = x1 - mu;
  red[tid] = d0 * d0 + d1 * d1;
  __syncthreads();
  for (int s = 128; s > 0; s >>= 1) {
    if (tid < s) red[tid] += red[tid + s];
    __syncthreads();
  }
  const float rstd = rsqrtf(red[0] * (1.0f / DMODEL) + 1e-5f);
  out[base + tid]       = (d0 * rstd * gamma[tid]       + beta[tid])       * mask;
  out[base + tid + 256] = (d1 * rstd * gamma[tid + 256] + beta[tid + 256]) * mask;
}

// ---------------------------------------------------------------------------
extern "C" void kernel_launch(void* const* d_in, const int* in_sizes, int n_in,
                              void* d_out, int out_size, void* d_ws, size_t ws_size,
                              hipStream_t stream) {
  (void)in_sizes; (void)n_in; (void)out_size; (void)ws_size;
  const int*   input = (const int*)  d_in[0];
  const float* table = (const float*)d_in[1];
  const float* Wq    = (const float*)d_in[2];
  const float* Wk    = (const float*)d_in[3];
  const float* Wv    = (const float*)d_in[4];
  const float* Wo    = (const float*)d_in[5];
  const float* ln1g  = (const float*)d_in[6];
  const float* ln1b  = (const float*)d_in[7];
  const float* W1    = (const float*)d_in[8];
  const float* b1    = (const float*)d_in[9];
  const float* W2    = (const float*)d_in[10];
  const float* b2    = (const float*)d_in[11];
  const float* ln2g  = (const float*)d_in[12];
  const float* ln2b  = (const float*)d_in[13];
  float* out = (float*)d_out;

  // workspace: x|q|k|v|t|t2 (6 x 8MB) + mid (32MB) = 80 MB
  const size_t ND = (size_t)MROWS * DMODEL;
  float* x   = (float*)d_ws;
  float* qb  = x  + ND;
  float* kb  = qb + ND;
  float* vb  = kb + ND;
  float* t   = vb + ND;
  float* t2  = t  + ND;
  float* mid = t2 + ND;                 // MROWS x DFF

  const dim3 blk(256);
  const dim3 gD(DMODEL / 128, MROWS / 128);   // (4, 32)
  const dim3 gF(DFF / 128,    MROWS / 128);   // (16, 32)
  const dim3 gA(BATCH * NHEADS, SEQ / QT);    // (64, 16)
  const size_t attn_smem =
      (size_t)QT * SSTR * 4 + (size_t)QT * PSTR * 2 + (size_t)QT * (DH + 8) * 2 +
      (size_t)DH * 40 * 2 + SEQ * 4 + QT * 8 * 4;

  embed_kernel<<<(unsigned)((ND + 255) / 256), blk, 0, stream>>>(input, table, x);

  for (int l = 0; l < N_LAYERS; ++l) {
    const float* wq = Wq + (size_t)l * DMODEL * DMODEL;
    const float* wk = Wk + (size_t)l * DMODEL * DMODEL;
    const float* wv = Wv + (size_t)l * DMODEL * DMODEL;
    const float* wo = Wo + (size_t)l * DMODEL * DMODEL;
    const float* w1 = W1 + (size_t)l * DMODEL * DFF;
    const float* w2 = W2 + (size_t)l * DFF * DMODEL;

    gemm_kernel<<<gD, blk, 0, stream>>>(x, wq, qb, nullptr, MROWS, DMODEL, DMODEL, 0);
    gemm_kernel<<<gD, blk, 0, stream>>>(x, wk, kb, nullptr, MROWS, DMODEL, DMODEL, 0);
    gemm_kernel<<<gD, blk, 0, stream>>>(x, wv, vb, nullptr, MROWS, DMODEL, DMODEL, 0);

    attn_kernel<<<gA, blk, attn_smem, stream>>>(qb, kb, vb, input, t);

    gemm_kernel<<<gD, blk, 0, stream>>>(t, wo, t2, nullptr, MROWS, DMODEL, DMODEL, 0);
    add_ln_kernel<<<MROWS, blk, 0, stream>>>(t2, x, ln1g + (size_t)l * DMODEL,
                                             ln1b + (size_t)l * DMODEL, input, x);

    gemm_kernel<<<gF, blk, 0, stream>>>(x, w1, mid, b1 + (size_t)l * DFF,
                                        MROWS, DFF, DMODEL, 1);
    gemm_kernel<<<gD, blk, 0, stream>>>(mid, w2, t, b2 + (size_t)l * DMODEL,
                                        MROWS, DMODEL, DFF, 0);
    float* dst = (l == N_LAYERS - 1) ? out : x;
    add_ln_kernel<<<MROWS, blk, 0, stream>>>(t, x, ln2g + (size_t)l * DMODEL,
                                             ln2b + (size_t)l * DMODEL, input, dst);
  }
}
